// RNN_18923625906935
// MI455X (gfx1250) — matrix operand order repaired and checked by reading, not compiled, not run on hardware
//
#include <hip/hip_runtime.h>
#include <hip/hip_bf16.h>

// ---------- CDNA5 WMMA types (gfx1250, wave32) ----------
typedef __attribute__((ext_vector_type(16))) __bf16   v16bf;
typedef __attribute__((ext_vector_type(8)))  float    v8f;
typedef __attribute__((ext_vector_type(4)))  unsigned v4u;

union AFrag { v16bf v; v4u q[2]; };  // A: two 16B chunks per lane (ISA 7.12.2)

__device__ __forceinline__ v8f wmma_bf16(v16bf a, v16bf b, v8f c) {
  // D = A(16x32 bf16) x B(32x16 bf16) + C(16x16 f32)
  return __builtin_amdgcn_wmma_f32_16x16x32_bf16(false, a, false, b,
                                                 (short)0, c, false, false);
}

// ---------- Kernel 0: f32 -> bf16 conversion ----------
__global__ void cvt_f32_bf16(const float* __restrict__ s, __bf16* __restrict__ d, int n) {
  int i = blockIdx.x * blockDim.x + threadIdx.x;
  int stride = gridDim.x * blockDim.x;
  for (; i < n; i += stride) d[i] = (__bf16)s[i];
}

// ---------- Kernel 1: sequential GRU scan (1 workgroup, 16 waves) ----------
// h-state is ping-ponged between two LDS buffers per step, so within a
// layer-step all reads hit the "old" buffer and all writes the "new" one:
// no read/write barrier, no spilled result registers.
__global__ __launch_bounds__(512)
void gru_scan(const int* __restrict__ sent, const float* __restrict__ wordvec,
              const __bf16* __restrict__ Wih, const __bf16* __restrict__ Whh,
              const float* __restrict__ bih, const float* __restrict__ bhh,
              const float* __restrict__ lng, const float* __restrict__ lnb,
              __bf16* __restrict__ Hout) {
  __shared__ __bf16 sH[2][2][32 * 256];  // [pingpong][layer]  64KB
  __shared__ __bf16 sHid[32 * 256];      // layer input        16KB
  __shared__ __bf16 sXh[32 * 256];       // LayerNorm output   16KB

  const int tid  = threadIdx.x;
  const int lane = tid & 31;
  const int wave = tid >> 5;             // 16 waves

  for (int i = tid; i < 4 * 8192; i += 512) ((__bf16*)sH)[i] = (__bf16)0.f;
  __syncthreads();

  for (int t = 0; t < 63; ++t) {
    const int rb = t & 1, wb = rb ^ 1;   // ping-pong h buffers

    // ---- embedding gather: x_t = wordvec[sent[b, t]] ----
    for (int i = tid; i < 8192; i += 512) {
      int b = i >> 8, k = i & 255;
      int tok = sent[b * 64 + t];
      sHid[i] = (__bf16)wordvec[tok * 256 + k];
    }
    __syncthreads();

    for (int j = 0; j < 2; ++j) {
      // ---- LayerNorm: one wave per row, 2 rows per wave (shuffle reduce) ----
      for (int rr = 0; rr < 2; ++rr) {
        int row = wave + rr * 16;
        float vals[8], s = 0.f;
        #pragma unroll
        for (int q = 0; q < 8; ++q) { vals[q] = (float)sHid[row * 256 + lane + q * 32]; s += vals[q]; }
        #pragma unroll
        for (int m2 = 16; m2 >= 1; m2 >>= 1) s += __shfl_xor(s, m2, 32);
        float mean = s * (1.f / 256.f);
        float vs = 0.f;
        #pragma unroll
        for (int q = 0; q < 8; ++q) { float d = vals[q] - mean; vs += d * d; }
        #pragma unroll
        for (int m2 = 16; m2 >= 1; m2 >>= 1) vs += __shfl_xor(vs, m2, 32);
        float inv = rsqrtf(vs * (1.f / 256.f) + 1e-5f);
        #pragma unroll
        for (int q = 0; q < 8; ++q) {
          int k = lane + q * 32;
          sXh[row * 256 + k] = (__bf16)((vals[q] - mean) * inv * lng[j * 256 + k] + lnb[j * 256 + k]);
        }
      }
      __syncthreads();

      // ---- gate GEMMs + fused gating + writes; 32 units over 16 waves ----
      for (int uu = 0; uu < 2; ++uu) {
        int unit = wave + uu * 16;
        int mt = unit >> 4, cg = unit & 15;
        int col = cg * 16 + (lane & 15);         // gate column (same across r/z/n)
        int kA = (lane >> 4) * 16;               // A-fragment k base (contiguous k, matches B)
        int kB = (lane >> 4) * 16;               // B-fragment k base
        const __bf16* ArX = sXh       + (mt * 16 + (lane & 15)) * 256;
        const __bf16* ArH = sH[rb][j] + (mt * 16 + (lane & 15)) * 256;
        v8f aR = {}, aZ = {}, aN1 = {}, aN2 = {};
        #pragma unroll
        for (int pass = 0; pass < 2; ++pass) {
          const __bf16* Ar = pass ? ArH : ArX;
          const __bf16* W  = (pass ? Whh : Wih) + j * 768 * 256;
          #pragma unroll 2
          for (int ks = 0; ks < 8; ++ks) {
            AFrag a;
            const __bf16* ap = Ar + ks * 32 + kA;
            a.q[0] = *(const v4u*)ap;            // ds_load_b128 x2
            a.q[1] = *(const v4u*)(ap + 8);
            v16bf b0 = *(const v16bf*)(W + (0 * 256 + col) * 256 + ks * 32 + kB);
            v16bf b1 = *(const v16bf*)(W + (1 * 256 + col) * 256 + ks * 32 + kB);
            v16bf b2 = *(const v16bf*)(W + (2 * 256 + col) * 256 + ks * 32 + kB);
            aR = wmma_bf16(a.v, b0, aR);         // r-gate (x and h chained)
            aZ = wmma_bf16(a.v, b1, aZ);         // z-gate
            if (pass == 0) aN1 = wmma_bf16(a.v, b2, aN1);  // i_n
            else           aN2 = wmma_bf16(a.v, b2, aN2);  // h_n
          }
        }
        float bR  = bih[j * 768 + col]       + bhh[j * 768 + col];
        float bZ  = bih[j * 768 + 256 + col] + bhh[j * 768 + 256 + col];
        float bNi = bih[j * 768 + 512 + col];
        float bNh = bhh[j * 768 + 512 + col];
        int m0 = mt * 16 + (lane >> 4) * 8;      // C/D layout: m = 8*(lane>>4)+i
        #pragma unroll
        for (int q = 0; q < 8; ++q) {
          int m = m0 + q;
          float hold = (float)sH[rb][j][m * 256 + col];
          float r = 1.f / (1.f + __expf(-(aR[q] + bR)));
          float z = 1.f / (1.f + __expf(-(aZ[q] + bZ)));
          float n = tanhf(aN1[q] + bNi + r * (aN2[q] + bNh));
          float hn = (1.f - z) * n + z * hold;
          float hid = hn + (float)sXh[m * 256 + col];      // residual
          sH[wb][j][m * 256 + col] = (__bf16)hn;           // new h (other buffer)
          if (j == 0) sHid[m * 256 + col] = (__bf16)hid;   // dead until next LN
          else        Hout[(t * 32 + m) * 256 + col] = (__bf16)hid;
        }
      }
      __syncthreads();   // hid/Xh handoff to next layer's LN
    }
  }
}

// ---------- Kernel 2: output projection [2016,256] x [256,32000] ----------
// 8 waves/WG, wave computes 16x80 tile; W bf16 is L2-resident (16MB << 192MB).
__global__ __launch_bounds__(256)
void logits_gemm(const __bf16* __restrict__ Hbf, const __bf16* __restrict__ Wbf,
                 const float* __restrict__ lb, float* __restrict__ out) {
  const int lane = threadIdx.x & 31;
  const int wave = threadIdx.x >> 5;
  const int mt = blockIdx.y;                       // 126 M-tiles
  const int nbase = blockIdx.x * 640 + wave * 80;  // 50 * 640 = 32000
  const int kA = (lane >> 4) * 16;
  const int kB = (lane >> 4) * 16;
  const __bf16* Arow = Hbf + (mt * 16 + (lane & 15)) * 256;

  v8f acc[5] = {};
  #pragma unroll
  for (int ks = 0; ks < 8; ++ks) {
    AFrag a;
    const __bf16* ap = Arow + ks * 32 + kA;
    a.q[0] = *(const v4u*)ap;
    a.q[1] = *(const v4u*)(ap + 8);
    if (ks < 7)  // warm next k-slice of weights (global_prefetch_b8)
      __builtin_prefetch(Wbf + (nbase + (lane & 15)) * 256 + (ks + 1) * 32 + kB, 0, 0);
    #pragma unroll
    for (int tt = 0; tt < 5; ++tt) {
      int ncol = nbase + tt * 16 + (lane & 15);
      v16bf b = *(const v16bf*)(Wbf + ncol * 256 + ks * 32 + kB);
      acc[tt] = wmma_bf16(a.v, b, acc[tt]);
    }
  }
  const int m0 = mt * 16 + (lane >> 4) * 8;
  #pragma unroll
  for (int tt = 0; tt < 5; ++tt) {
    int ncol = nbase + tt * 16 + (lane & 15);
    float bias = lb[ncol];
    #pragma unroll
    for (int q = 0; q < 8; ++q) {
      int m = m0 + q;                      // m = t*32 + b
      int tt2 = m >> 5, b = m & 31;
      out[1 + (size_t)(b * 63 + tt2) * 32000 + ncol] = acc[tt][q] + bias;
    }
  }
}

// ---------- Kernel 3: per-row online logsumexp + masked CE ----------
__global__ __launch_bounds__(256)
void ce_rows(const float* __restrict__ out, const int* __restrict__ sent,
             float* __restrict__ rowloss) {
  int m = blockIdx.x;                    // 2016 rows, m = t*32 + b
  int t = m >> 5, b = m & 31;
  const float* row = out + 1 + (size_t)(b * 63 + t) * 32000;
  float mx = -3.0e38f, sm = 0.f;
  for (int c = threadIdx.x; c < 32000; c += 256) {
    float v = row[c];
    if (v > mx) { sm = sm * __expf(mx - v) + 1.f; mx = v; }
    else        sm += __expf(v - mx);
  }
  __shared__ float sMx[256], sSm[256];
  sMx[threadIdx.x] = mx; sSm[threadIdx.x] = sm;
  __syncthreads();
  for (int off = 128; off >= 1; off >>= 1) {     // deterministic tree
    if (threadIdx.x < off) {
      float m1 = sMx[threadIdx.x],       s1 = sSm[threadIdx.x];
      float m2 = sMx[threadIdx.x + off], s2 = sSm[threadIdx.x + off];
      float mm = fmaxf(m1, m2);
      sMx[threadIdx.x] = mm;
      sSm[threadIdx.x] = s1 * __expf(m1 - mm) + s2 * __expf(m2 - mm);
    }
    __syncthreads();
  }
  if (threadIdx.x == 0) {
    int tgt = sent[b * 64 + t + 1];
    float lse = logf(sSm[0]) + sMx[0];
    float ce = lse - row[tgt];
    rowloss[m] = (tgt != 0) ? ce : 0.f;
  }
}

// ---------- Kernel 4: deterministic final reduction ----------
__global__ __launch_bounds__(256)
void finalize_loss(const float* __restrict__ rowloss, const int* __restrict__ length,
                   float* __restrict__ out) {
  __shared__ float red[256];
  float s = 0.f;
  for (int i = threadIdx.x; i < 2016; i += 256) s += rowloss[i];
  red[threadIdx.x] = s;
  __syncthreads();
  for (int off = 128; off >= 1; off >>= 1) {
    if (threadIdx.x < off) red[threadIdx.x] += red[threadIdx.x + off];
    __syncthreads();
  }
  if (threadIdx.x == 0) {
    int ls = 0;
    for (int i = 0; i < 32; ++i) ls += length[i];
    out[0] = red[0] / (float)ls;
  }
}

extern "C" void kernel_launch(void* const* d_in, const int* in_sizes, int n_in,
                              void* d_out, int out_size, void* d_ws, size_t ws_size,
                              hipStream_t stream) {
  (void)in_sizes; (void)n_in; (void)out_size; (void)ws_size;
  const int*   sent    = (const int*)  d_in[0];
  const int*   length  = (const int*)  d_in[1];
  const float* wordvec = (const float*)d_in[2];
  const float* W_ih    = (const float*)d_in[3];
  const float* W_hh    = (const float*)d_in[4];
  const float* b_ih    = (const float*)d_in[5];
  const float* b_hh    = (const float*)d_in[6];
  const float* ln_g    = (const float*)d_in[7];
  const float* ln_b    = (const float*)d_in[8];
  const float* lin_w   = (const float*)d_in[9];
  const float* lin_b   = (const float*)d_in[10];
  float* out = (float*)d_out;

  char* ws = (char*)d_ws;
  __bf16* Wout_bf = (__bf16*)(ws);                               // 16,384,000 B
  __bf16* Wih_bf  = (__bf16*)(ws + 16384000);                    //    786,432 B
  __bf16* Whh_bf  = (__bf16*)(ws + 16384000 + 786432);           //    786,432 B
  __bf16* Hid_bf  = (__bf16*)(ws + 16384000 + 2 * 786432);       //  1,032,192 B
  float*  rowloss = (float*) (ws + 16384000 + 2 * 786432 + 1032192); // 8,064 B

  cvt_f32_bf16<<<512, 256, 0, stream>>>(lin_w, Wout_bf, 32000 * 256);
  cvt_f32_bf16<<<96,  256, 0, stream>>>(W_ih,  Wih_bf,  2 * 768 * 256);
  cvt_f32_bf16<<<96,  256, 0, stream>>>(W_hh,  Whh_bf,  2 * 768 * 256);

  gru_scan<<<1, 512, 0, stream>>>(sent, wordvec, Wih_bf, Whh_bf,
                                  b_ih, b_hh, ln_g, ln_b, Hid_bf);

  logits_gemm<<<dim3(50, 126), 256, 0, stream>>>(Hid_bf, Wout_bf, lin_b, out);

  ce_rows<<<2016, 256, 0, stream>>>(out, sent, rowloss);
  finalize_loss<<<1, 256, 0, stream>>>(rowloss, length, out);
}
